// GAT_LSTM_Model_61469571940802
// MI455X (gfx1250) — matrix operand order repaired and checked by reading, not compiled, not run on hardware
//
#include <hip/hip_runtime.h>
#include <hip/hip_bf16.h>
#include <math.h>

// Problem constants (from reference)
#define NN  131072      // B*T nodes
#define TT  1024
#define BB  128
#define HH  4           // attention heads
#define EE  (NN*5)      // DEG*N + N self loops
#define HL  64

typedef __bf16 bf16_t;
typedef bf16_t v16bf __attribute__((ext_vector_type(16)));
typedef float  v8f   __attribute__((ext_vector_type(8)));

#if defined(__HIP_DEVICE_COMPILE__) && __has_builtin(__builtin_amdgcn_global_load_async_to_lds_b128)
#define HAVE_ASYNC 1
#else
#define HAVE_ASYNC 0
#endif

#if HAVE_ASYNC
static __device__ __forceinline__ void async_b128(const float* gp, float* lp) {
  typedef int v4i_t __attribute__((__vector_size__(4 * sizeof(int))));
  typedef __attribute__((address_space(1))) v4i_t* gp_t;
  typedef __attribute__((address_space(3))) v4i_t* lp_t;
  __builtin_amdgcn_global_load_async_to_lds_b128((gp_t)(void*)gp, (lp_t)(void*)lp, 0, 0);
}
static __device__ __forceinline__ void wait_async0() {
#if __has_builtin(__builtin_amdgcn_s_wait_asynccnt)
  __builtin_amdgcn_s_wait_asynccnt(0);
#else
  asm volatile("s_wait_asynccnt 0" ::: "memory");
#endif
}
#endif

static __device__ __forceinline__ bf16_t f2bf(float f) {
  unsigned u = __builtin_bit_cast(unsigned, f);
  unsigned r = (u + 0x7fffu + ((u >> 16) & 1u)) >> 16;   // RNE
  return __builtin_bit_cast(bf16_t, (unsigned short)r);
}
static __device__ __forceinline__ v8f wmma_bf16(v16bf a, v16bf b, v8f c) {
  // D = A(16x32 bf16) x B(32x16 bf16) + C(16x16 f32)
  return __builtin_amdgcn_wmma_f32_16x16x32_bf16(false, a, false, b, (short)0, c,
                                                 false, false);
}
static __device__ __forceinline__ float sigm(float x) { return 1.f / (1.f + __expf(-x)); }

// order-preserving float<->uint encoding for atomic max
static __device__ __forceinline__ unsigned fenc(float f) {
  unsigned u = __builtin_bit_cast(unsigned, f);
  return u ^ ((unsigned)((int)u >> 31) | 0x80000000u);
}
static __device__ __forceinline__ float fdec(unsigned e) {
  unsigned u = (e & 0x80000000u) ? (e ^ 0x80000000u) : ~e;
  return __builtin_bit_cast(float, u);
}

// ---------------------------------------------------------------- utility
__global__ __launch_bounds__(256) void k_fillf(float* p, float v, long n) {
  long i = (long)blockIdx.x * blockDim.x + threadIdx.x;
  if (i < n) p[i] = v;
}
__global__ __launch_bounds__(256) void k_rowbias(float* p, const float* __restrict__ bias,
                                                 long rows, int C) {
  long i = (long)blockIdx.x * blockDim.x + threadIdx.x;
  if (i < rows * (long)C) p[i] = bias[i % C];
}

// ---------------------------------------------------------------- feature assembly
__global__ __launch_bounds__(256) void k_build_x(
    const float* gnd, const float* root, const float* chg, const float* num,
    const float* stp, const float* brn, const int* bidx, const int* tidx,
    const int* iidx, const int* sidx, const int* oidx, const float* eb,
    const float* et, const float* eid, const float* es, const float* eo, float* x) {
  long n = (long)blockIdx.x * blockDim.x + threadIdx.x;
  if (n >= NN) return;
  float* o = x + n * 24;
  o[0] = gnd[n * 2]; o[1] = gnd[n * 2 + 1];
  o[2] = eb[bidx[n]];
  o[3] = root[n * 2]; o[4] = root[n * 2 + 1];
  o[5] = chg[n * 2]; o[6] = chg[n * 2 + 1];
  int ti = tidx[n]; o[7] = et[ti * 2]; o[8] = et[ti * 2 + 1];
  int ii = iidx[n]; ii = (ii < 700) ? ii : ((ii - 700) % 300 + 700);
  o[9] = eid[ii * 4]; o[10] = eid[ii * 4 + 1]; o[11] = eid[ii * 4 + 2]; o[12] = eid[ii * 4 + 3];
  int si = sidx[n]; si = (si < 700) ? si : ((si - 700) % 300 + 700);
  o[13] = es[si * 4]; o[14] = es[si * 4 + 1]; o[15] = es[si * 4 + 2]; o[16] = es[si * 4 + 3];
  o[17] = num[n];
  int op = oidx[n]; o[18] = eo[op * 2]; o[19] = eo[op * 2 + 1];
  o[20] = stp[n];
  o[21] = brn[n * 3]; o[22] = brn[n * 3 + 1]; o[23] = brn[n * 3 + 2];
}

// ---------------------------------------------------------------- generic bf16 WMMA GEMM
// C[M x Nc] = A[M x K] * (BT ? B^T : B) + bias ; A row-major (lda=K),
// B row-major KxNc, or (BT) row-major NcxK. One 16x16 tile per wave.
// The block's K x 16 B-strip is staged once into LDS via async copies
// (ASYNCcnt path) when the tile is full and K fits; fragments then come
// from LDS instead of strided global loads.
template <int BT>
__global__ __launch_bounds__(256) void k_gemm(const float* __restrict__ A,
                                              const float* __restrict__ Bm,
                                              const float* __restrict__ bias,
                                              float* __restrict__ C,
                                              int M, int Nc, int K, int ldc) {
  __shared__ float bpanel[128 * 16];     // up to 8 KB staged B strip
  const int lane = threadIdx.x & 31;
  const int wv = threadIdx.x >> 5;
  const int mt = blockIdx.y * 8 + wv;
  const int nt = blockIdx.x;
  const int mBase = mt * 16, nBase = nt * 16;
  const int m = mBase + (lane & 15);
  const int n = nBase + (lane & 15);

  const bool useP = (nBase + 16 <= Nc) && (K <= 128) && ((K & 3) == 0);
  if (useP) {
    // LDS layout: BT -> [r][k] (r*K + k), else [k][c] (k*16 + c); both match
    // a linear 16-byte-chunk fill.
    const int chunks = K * 4;            // 16B chunks in the panel
#if HAVE_ASYNC
    if (BT) {
      const int cpr = K >> 2;            // 16B chunks per K-float row
      for (int c = threadIdx.x; c < chunks; c += 256) {
        const int r = c / cpr;
        const float* gp = Bm + (long)(nBase + r) * K + ((c % cpr) << 2);
        async_b128(gp, bpanel + (c << 2));
      }
    } else {
      for (int c = threadIdx.x; c < chunks; c += 256) {
        const float* gp = Bm + (long)(c >> 2) * Nc + nBase + ((c & 3) << 2);
        async_b128(gp, bpanel + (c << 2));
      }
    }
    wait_async0();
#else
    for (int i = threadIdx.x; i < K * 16; i += 256) {
      if (BT) { int r = i / K, k = i % K; bpanel[i] = Bm[(long)(nBase + r) * K + k]; }
      else    { int k = i >> 4, cc = i & 15; bpanel[i] = Bm[(long)k * Nc + nBase + cc]; }
    }
#endif
    __syncthreads();
  }
  if (mt * 16 >= M) return;

  v8f acc = {0.f, 0.f, 0.f, 0.f, 0.f, 0.f, 0.f, 0.f};
  const int kSteps = (K + 31) / 32;
  for (int ks = 0; ks < kSteps; ++ks) {
    const int kBase = ks * 32;
    v16bf af;
    {
      const float* row = A + (long)m * K;
      const int k0 = kBase + ((lane >> 4) << 4);
#pragma unroll
      for (int i = 0; i < 8; ++i) {
        int ka = k0 + i, kb = k0 + 8 + i;
        af[i] = f2bf(ka < K ? row[ka] : 0.f);
        af[i + 8] = f2bf(kb < K ? row[kb] : 0.f);
      }
    }
    v16bf bfr;
    {
      const int kb0 = kBase + ((lane >> 4) << 4);
      if (useP) {
        if (BT) {
          const float* br = bpanel + (long)(lane & 15) * K;
#pragma unroll
          for (int i = 0; i < 16; ++i) {
            int k = kb0 + i;
            bfr[i] = f2bf(k < K ? br[k] : 0.f);
          }
        } else {
#pragma unroll
          for (int i = 0; i < 16; ++i) {
            int k = kb0 + i;
            bfr[i] = f2bf(k < K ? bpanel[k * 16 + (lane & 15)] : 0.f);
          }
        }
      } else if (BT) {
        const float* rowb = Bm + (long)n * K;
#pragma unroll
        for (int i = 0; i < 16; ++i) {
          int k = kb0 + i;
          bfr[i] = f2bf((k < K && n < Nc) ? rowb[k] : 0.f);
        }
      } else {
#pragma unroll
        for (int i = 0; i < 16; ++i) {
          int k = kb0 + i;
          bfr[i] = f2bf((k < K && n < Nc) ? Bm[(long)k * Nc + n] : 0.f);
        }
      }
    }
    acc = wmma_bf16(af, bfr, acc);
  }
  if (n < Nc) {
    float bv = bias ? bias[n] : 0.f;
    const int mOff = (lane >> 4) << 3;
#pragma unroll
    for (int r = 0; r < 8; ++r)
      C[(long)(mBase + mOff + r) * ldc + n] = acc[r] + bv;
  }
}

// ---------------------------------------------------------------- GAT attention pieces
__global__ __launch_bounds__(256) void k_att(const float* __restrict__ h,
                                             const float* __restrict__ asrc,
                                             const float* __restrict__ adst,
                                             float* as_, float* ad_, int C) {
  long id = (long)blockIdx.x * blockDim.x + threadIdx.x;
  if (id >= (long)NN * HH) return;
  int n = (int)(id / HH), hh = (int)(id % HH);
  const float* row = h + (long)n * (HH * C) + hh * C;
  const float* ws = asrc + hh * C;
  const float* wd = adst + hh * C;
  float s = 0.f, d = 0.f;
  for (int c = 0; c < C; ++c) { s += row[c] * ws[c]; d += row[c] * wd[c]; }
  as_[id] = s; ad_[id] = d;
}

__global__ __launch_bounds__(256) void k_edge_max(const int* __restrict__ src,
                                                  const int* __restrict__ dst,
                                                  const float* __restrict__ as_,
                                                  const float* __restrict__ ad_,
                                                  float* ebuf, unsigned* mx) {
  long id = (long)blockIdx.x * blockDim.x + threadIdx.x;
  if (id >= (long)EE * HH) return;
  long e = id / HH; int hh = (int)(id % HH);
  int s = src[e], d = dst[e];
  float v = as_[(long)s * HH + hh] + ad_[(long)d * HH + hh];
  v = v > 0.f ? v : 0.2f * v;           // leaky_relu(0.2)
  ebuf[id] = v;
  atomicMax(&mx[(long)d * HH + hh], fenc(v));
}

__global__ __launch_bounds__(256) void k_edge_exp(const int* __restrict__ dst,
                                                  float* ebuf, const unsigned* __restrict__ mx,
                                                  float* sm) {
  long id = (long)blockIdx.x * blockDim.x + threadIdx.x;
  if (id >= (long)EE * HH) return;
  long e = id / HH; int hh = (int)(id % HH);
  int d = dst[e];
  float ex = __expf(ebuf[id] - fdec(mx[(long)d * HH + hh]));
  ebuf[id] = ex;
  atomicAdd(&sm[(long)d * HH + hh], ex);
}

__global__ __launch_bounds__(256) void k_scatter(const int* __restrict__ src,
                                                 const int* __restrict__ dst,
                                                 const float* __restrict__ ebuf,
                                                 const float* __restrict__ sm,
                                                 const float* __restrict__ feat,
                                                 float* out, int C, int ldc,
                                                 int headStride, float scale) {
  long id = (long)blockIdx.x * blockDim.x + threadIdx.x;
  if (id >= (long)EE * HH) return;
  long e = id / HH; int hh = (int)(id % HH);
  int s = src[e], d = dst[e];
  float alpha = ebuf[id] / (sm[(long)d * HH + hh] + 1e-16f) * scale;
  const float* f = feat + (long)s * (HH * C) + hh * C;
  float* o = out + (long)d * ldc + hh * headStride;
  for (int c = 0; c < C; ++c) atomicAdd(&o[c], alpha * f[c]);
}

// ---------------------------------------------------------------- persistent BiLSTM scan
// grid = 2 (dir), block = 256 (8 waves). Wave w owns batch rows [16w,16w+16).
// Whh (256x64) staged as bf16 in LDS; h kept per-wave in LDS (bf16, A-layout
// friendly row-major 16x64); c kept in registers. 32 WMMAs / wave / step.
// No inter-wave barriers inside the 1024-step scan.
__global__ __launch_bounds__(256) void k_lstm(const float* __restrict__ xgF,
                                              const float* __restrict__ xgB,
                                              const float* __restrict__ Whh,
                                              const float* __restrict__ bhh,
                                              float* __restrict__ outSeq, int ldo) {
  __shared__ bf16_t whh_s[256 * 64];     // B^T source: (n,k) row-major, 32 KB
  __shared__ bf16_t hbuf[8][16 * 64];    // per-wave h_prev, 16 KB total
  const int dir = blockIdx.x;
  const float* xg = dir ? xgB : xgF;
  const float* W = Whh + (long)dir * 256 * 64;
  const float* bh = bhh + dir * 256;
  const int colOff = dir * HL;
  const int lane = threadIdx.x & 31;
  const int wv = threadIdx.x >> 5;
  const int l15 = lane & 15;
  const int mOff = (lane >> 4) << 3;

  for (int i = threadIdx.x; i < 256 * 64; i += 256) whh_s[i] = f2bf(W[i]);
  for (int i = lane; i < 16 * 64; i += 32) hbuf[wv][i] = f2bf(0.f);
  __syncthreads();

  v8f cfr[4];
#pragma unroll
  for (int jt = 0; jt < 4; ++jt) cfr[jt] = (v8f){0.f, 0.f, 0.f, 0.f, 0.f, 0.f, 0.f, 0.f};

  for (int s = 0; s < TT; ++s) {
    const int t = dir ? (TT - 1 - s) : s;
    // prefetch next timestep's gate-input row (global_prefetch_b8)
    {
      const int tn = dir ? (t - 1) : (t + 1);
      if (tn >= 0 && tn < TT)
        __builtin_prefetch(xg + ((long)(wv * 16 + mOff) * TT + tn) * 256 + lane * 8, 0, 1);
    }
    // A fragments of h_prev (K = 0..63), wave-private LDS
    v16bf a0, a1;
    {
      const bf16_t* hr = &hbuf[wv][l15 * 64];
      const int k0 = (lane >> 4) << 4;
#pragma unroll
      for (int i = 0; i < 8; ++i) {
        a0[i] = hr[k0 + i];       a0[i + 8] = hr[k0 + 8 + i];
        a1[i] = hr[32 + k0 + i];  a1[i + 8] = hr[32 + k0 + 8 + i];
      }
    }
    v8f acc[4][4];   // [gate][jt]
#pragma unroll
    for (int jt = 0; jt < 4; ++jt) {
#pragma unroll
      for (int gi = 0; gi < 4; ++gi) {
        const int nb = gi * HL + jt * 16;
        const int nn = nb + l15;
        v16bf b0, b1;
        {
          const bf16_t* br = &whh_s[nn * 64];
          const int kb0 = (lane >> 4) << 4;
#pragma unroll
          for (int i = 0; i < 16; ++i) { b0[i] = br[kb0 + i]; b1[i] = br[32 + kb0 + i]; }
        }
        v8f a = {0.f, 0.f, 0.f, 0.f, 0.f, 0.f, 0.f, 0.f};
        a = wmma_bf16(a0, b0, a);
        a = wmma_bf16(a1, b1, a);
        const float bias = bh[nn];
        const float* xc = xg + ((long)(wv * 16 + mOff) * TT + t) * 256 + nn;
#pragma unroll
        for (int r = 0; r < 8; ++r) a[r] += xc[(long)r * TT * 256] + bias;
        acc[gi][jt] = a;
      }
    }
    // elementwise gate mixing; write h to outSeq + wave-private LDS
#pragma unroll
    for (int jt = 0; jt < 4; ++jt) {
      v8f c = cfr[jt];
#pragma unroll
      for (int r = 0; r < 8; ++r) {
        float i_ = sigm(acc[0][jt][r]);
        float f_ = sigm(acc[1][jt][r]);
        float g_ = tanhf(acc[2][jt][r]);
        float o_ = sigm(acc[3][jt][r]);
        float cn = f_ * c[r] + i_ * g_;
        float hn = o_ * tanhf(cn);
        c[r] = cn;
        const int row = wv * 16 + mOff + r;
        outSeq[((long)row * TT + t) * ldo + colOff + jt * 16 + l15] = hn;
        hbuf[wv][(mOff + r) * 64 + jt * 16 + l15] = f2bf(hn);
      }
      cfr[jt] = c;
    }
  }
}

__global__ __launch_bounds__(256) void k_last(const float* __restrict__ seq, float* last) {
  int id = blockIdx.x * blockDim.x + threadIdx.x;
  if (id >= BB * 128) return;
  int b = id >> 7, c = id & 127;
  last[id] = seq[((long)b * TT + (TT - 1)) * 128 + c];
}

// ---------------------------------------------------------------- launcher
extern "C" void kernel_launch(void* const* d_in, const int* in_sizes, int n_in,
                              void* d_out, int out_size, void* d_ws, size_t ws_size,
                              hipStream_t stream) {
  (void)in_sizes; (void)n_in; (void)out_size; (void)ws_size;
  const float* grounded   = (const float*)d_in[0];
  const float* root       = (const float*)d_in[1];
  const float* changeable = (const float*)d_in[2];
  const float* number     = (const float*)d_in[3];
  const float* step_in    = (const float*)d_in[4];
  const float* branch     = (const float*)d_in[5];
  const int*   bound_idx  = (const int*)d_in[6];
  const int*   type_idx   = (const int*)d_in[7];
  const int*   ident_idx  = (const int*)d_in[8];
  const int*   string_idx = (const int*)d_in[9];
  const int*   op_idx     = (const int*)d_in[10];
  const int*   ei         = (const int*)d_in[11];
  const float* emb_bound  = (const float*)d_in[12];
  const float* emb_type   = (const float*)d_in[13];
  const float* emb_ident  = (const float*)d_in[14];
  const float* emb_string = (const float*)d_in[15];
  const float* emb_op     = (const float*)d_in[16];
  const float* W1         = (const float*)d_in[17];
  const float* att_src1   = (const float*)d_in[18];
  const float* att_dst1   = (const float*)d_in[19];
  const float* b1         = (const float*)d_in[20];
  const float* W2         = (const float*)d_in[21];
  const float* att_src2   = (const float*)d_in[22];
  const float* att_dst2   = (const float*)d_in[23];
  const float* b2         = (const float*)d_in[24];
  const float* Wih0       = (const float*)d_in[25];
  const float* Wih_rest   = (const float*)d_in[26];
  const float* Whh        = (const float*)d_in[27];
  const float* bih        = (const float*)d_in[28];
  const float* bhh        = (const float*)d_in[29];
  const float* fc1_W      = (const float*)d_in[30];
  const float* fc1_b      = (const float*)d_in[31];
  const float* head_W     = (const float*)d_in[32];
  const float* head_b     = (const float*)d_in[33];
  float* outF = (float*)d_out;

  const int* srcp = ei;
  const int* dstp = ei + EE;

  // bump-allocated, lifetime-reused workspace (~436 MB)
  float* ws0 = (float*)d_ws;                 // N*64
  float* ws1 = ws0 + (size_t)NN * 64;        // N*128
  float* ws2 = ws1 + (size_t)NN * 128;       // N*128
  float* ws3 = ws2 + (size_t)NN * 128;       // N*256
  float* ws4 = ws3 + (size_t)NN * 256;       // N*256

  float* x    = ws1;                         // N*24
  float* h1   = ws3;                         // N*96
  float* as_  = ws4;                         // N*4
  float* ad_  = ws4 + (size_t)NN * HH;
  unsigned* mx = (unsigned*)(ws4 + (size_t)2 * NN * HH);
  float* sm   = ws4 + (size_t)3 * NN * HH;
  float* ebuf = ws4 + (size_t)4 * NN * HH;   // E*4
  float* g1   = ws2;                         // N*96
  float* h2   = ws3;                         // N*256 (h1 dead)
  float* g2   = ws0;                         // N*64
  float* xgF  = ws3;                         // N*256 (h2 dead)
  float* xgB  = ws4;                         // N*256 (GAT scratch dead)
  float* last = ws0;                         // 128*128 (g2 dead)
  float* lin  = ws0 + 128 * 128;             // 128*64

  dim3 blk(256);
  const long nh = (long)NN * HH;
  const long eh = (long)EE * HH;

  // 1) feature assembly
  k_build_x<<<(NN + 255) / 256, blk, 0, stream>>>(grounded, root, changeable, number,
      step_in, branch, bound_idx, type_idx, ident_idx, string_idx, op_idx,
      emb_bound, emb_type, emb_ident, emb_string, emb_op, x);

  // 2) GAT layer 1: h1 = x @ W1 (N x 24 -> N x 96)
  {
    dim3 g(96 / 16, (NN / 16 + 7) / 8);
    k_gemm<0><<<g, blk, 0, stream>>>(x, W1, nullptr, h1, NN, 96, 24, 96);
  }
  k_att<<<(unsigned)((nh + 255) / 256), blk, 0, stream>>>(h1, att_src1, att_dst1, as_, ad_, 24);
  k_fillf<<<(unsigned)((2 * nh + 255) / 256), blk, 0, stream>>>((float*)mx, 0.f, 2 * nh);
  k_edge_max<<<(unsigned)((eh + 255) / 256), blk, 0, stream>>>(srcp, dstp, as_, ad_, ebuf, mx);
  k_edge_exp<<<(unsigned)((eh + 255) / 256), blk, 0, stream>>>(dstp, ebuf, mx, sm);
  k_rowbias<<<(unsigned)(((long)NN * 96 + 255) / 256), blk, 0, stream>>>(g1, b1, NN, 96);
  k_scatter<<<(unsigned)((eh + 255) / 256), blk, 0, stream>>>(srcp, dstp, ebuf, sm, h1, g1,
                                                              24, 96, 24, 1.0f);

  // 3) GAT layer 2: h2 = g1 @ W2 (N x 96 -> N x 256), mean over heads -> g2 (N x 64)
  {
    dim3 g(256 / 16, (NN / 16 + 7) / 8);
    k_gemm<0><<<g, blk, 0, stream>>>(g1, W2, nullptr, h2, NN, 256, 96, 256);
  }
  k_att<<<(unsigned)((nh + 255) / 256), blk, 0, stream>>>(h2, att_src2, att_dst2, as_, ad_, 64);
  k_fillf<<<(unsigned)((2 * nh + 255) / 256), blk, 0, stream>>>((float*)mx, 0.f, 2 * nh);
  k_edge_max<<<(unsigned)((eh + 255) / 256), blk, 0, stream>>>(srcp, dstp, as_, ad_, ebuf, mx);
  k_edge_exp<<<(unsigned)((eh + 255) / 256), blk, 0, stream>>>(dstp, ebuf, mx, sm);
  k_rowbias<<<(unsigned)(((long)NN * 64 + 255) / 256), blk, 0, stream>>>(g2, b2, NN, 64);
  k_scatter<<<(unsigned)((eh + 255) / 256), blk, 0, stream>>>(srcp, dstp, ebuf, sm, h2, g2,
                                                              64, 64, 0, 0.25f);

  // 4) 4 x BiLSTM
  const float* seqIn = g2;
  int Kin = 64;
  float* seqOut = ws1;
  for (int l = 0; l < 4; ++l) {
    const float* WihF;
    const float* WihB;
    if (l == 0) { WihF = Wih0; WihB = Wih0 + 256 * 64; }
    else {
      WihF = Wih_rest + (size_t)((l - 1) * 2) * 256 * 128;
      WihB = WihF + 256 * 128;
    }
    dim3 gx(256 / 16, (NN / 16 + 7) / 8);
    k_gemm<1><<<gx, blk, 0, stream>>>(seqIn, WihF, bih + (l * 2) * 256, xgF, NN, 256, Kin, 256);
    k_gemm<1><<<gx, blk, 0, stream>>>(seqIn, WihB, bih + (l * 2 + 1) * 256, xgB, NN, 256, Kin, 256);
    k_lstm<<<2, blk, 0, stream>>>(xgF, xgB, Whh + (size_t)(l * 2) * 256 * 64,
                                  bhh + (l * 2) * 256, seqOut, 128);
    seqIn = seqOut;
    Kin = 128;
    seqOut = (seqOut == ws1) ? ws2 : ws1;
  }
  // after 4 layers final sequence lives in ws2

  // 5) head MLP on last timestep
  k_last<<<(BB * 128 + 255) / 256, blk, 0, stream>>>(ws2, last);
  {
    dim3 g(64 / 16, 1);
    k_gemm<0><<<g, blk, 0, stream>>>(last, fc1_W, fc1_b, lin, 128, 64, 128, 64);
  }
  {
    dim3 g((70 + 15) / 16, 1);
    k_gemm<0><<<g, blk, 0, stream>>>(lin, head_W, head_b, outF, 128, 70, 64, 70);
  }
}
